// InterAttention_83769042141765
// MI455X (gfx1250) — matrix operand order repaired and checked by reading, not compiled, not run on hardware
//
#include <hip/hip_runtime.h>
#include <hip/hip_bf16.h>

// Problem constants (match reference): B=16, C=512, H=W=32, HW=1024
#define BB 16
#define CC 512
#define HWD 1024

// LDS row pitch for staged panels: 32 data halfs padded to 40 (80B) so that
// consecutive rows map to bank offsets k*20 mod 64 -> 16 disjoint 4-bank groups.
#define LPITCH 40

typedef __attribute__((ext_vector_type(16))) _Float16 v16h;
typedef __attribute__((ext_vector_type(8)))  _Float16 v8h;
typedef __attribute__((ext_vector_type(4)))  _Float16 v4h;
typedef __attribute__((ext_vector_type(8)))  float    v8f;

union FragU { v16h f; struct { v8h lo; v8h hi; } h; };

// ---------------- fragment loaders (global, row-major K-contiguous) ----------------

// A-fragment (16x32 f16, M x K): lane&15 = row M; lanes 0-15 hold K {0..7,16..23},
// lanes 16-31 hold K {8..15,24..31}  -> two b128 loads from a row-major M x K tile.
__device__ __forceinline__ v16h load_frag_a(const _Float16* __restrict__ p, int ld, int lane) {
  const int row = lane & 15, sel = lane >> 4;
  const _Float16* r = p + (size_t)row * ld + sel * 8;
  FragU u;
  u.h.lo = *(const v8h*)(r);
  u.h.hi = *(const v8h*)(r + 16);
  return u.f;
}

// B-fragment (32x16 f16, K x N): lane&15 = column N; lanes 0-15 hold K {0..7,16..23},
// lanes 16-31 hold K {8..15,24..31} -> two b128 loads from a row-major N x K tile (B^T rows).
__device__ __forceinline__ v16h load_frag_b(const _Float16* __restrict__ p, int ld, int lane) {
  const int row = lane & 15, sel = lane >> 4;
  const _Float16* r = p + (size_t)row * ld + sel * 8;
  FragU u;
  u.h.lo = *(const v8h*)(r);
  u.h.hi = *(const v8h*)(r + 16);
  return u.f;
}

// ---------------- fragment loaders (LDS panels, rows padded to LPITCH halfs) ----------------

__device__ __forceinline__ v16h lds_frag_b(const _Float16* lrow, int lane) {
  const int row = lane & 15, sel = lane >> 4;
  const _Float16* r = lrow + row * LPITCH + sel * 8;
  FragU u;
  u.h.lo = *(const v8h*)(r);
  u.h.hi = *(const v8h*)(r + 16);
  return u.f;
}

__device__ __forceinline__ v16h lds_frag_a(const _Float16* lt, int lane) {
  const int row = lane & 15, sel = lane >> 4;
  const _Float16* r = lt + row * LPITCH + sel * 8;
  FragU u;
  u.h.lo = *(const v8h*)(r);
  u.h.hi = *(const v8h*)(r + 16);
  return u.f;
}

// ---------------- async global -> LDS staging (ASYNCcnt path) ----------------

__device__ __forceinline__ void async_copy_b128(const _Float16* g, _Float16* l) {
  unsigned lds = (unsigned)(size_t)(const void*)l;       // low 32 bits = LDS offset
  unsigned long long ga = (unsigned long long)(size_t)g;
  asm volatile("global_load_async_to_lds_b128 %0, %1, off" :: "v"(lds), "v"(ga) : "memory");
}

__device__ __forceinline__ void wait_async0() {
  asm volatile("s_wait_asynccnt 0" ::: "memory");
}

// Stage a 128-row x 32-half panel (row-major, row stride ld) into LDS. 256 threads.
__device__ __forceinline__ void stage_panel128(const _Float16* __restrict__ g, int ld,
                                               _Float16* lbuf, int tid) {
#pragma unroll
  for (int i = 0; i < 2; ++i) {
    const int id = tid + i * 256;          // 0..511 : 128 rows x 4 chunks of 8 halfs
    const int row = id >> 2, c = id & 3;
    async_copy_b128(g + (size_t)row * ld + c * 8, lbuf + row * LPITCH + c * 8);
  }
}

// Stage a 16-row x 32-half tile into LDS. First 64 threads participate.
__device__ __forceinline__ void stage_tile16(const _Float16* __restrict__ g, int ld,
                                             _Float16* lbuf, int tid) {
  if (tid < 64) {
    const int row = tid >> 2, c = tid & 3;
    async_copy_b128(g + (size_t)row * ld + c * 8, lbuf + row * LPITCH + c * 8);
  }
}

// ---------------- Pre-pass: f32 -> f16 conversions ----------------

__global__ __launch_bounds__(256) void convert_f32_f16(const float* __restrict__ s,
                                                       _Float16* __restrict__ d, size_t n4) {
  size_t i = (size_t)blockIdx.x * blockDim.x + threadIdx.x;
  size_t stride = (size_t)gridDim.x * blockDim.x;
  for (; i < n4; i += stride) {
    float4 v = ((const float4*)s)[i];
    v4h o;
    o[0] = (_Float16)v.x; o[1] = (_Float16)v.y; o[2] = (_Float16)v.z; o[3] = (_Float16)v.w;
    ((v4h*)d)[i] = o;
  }
}

// src [B][CC][HWD] f32 -> dst [B][HWD][CC] f16 (LDS-tiled 32x32 transpose)
__global__ __launch_bounds__(256) void transpose_convert(const float* __restrict__ src,
                                                         _Float16* __restrict__ dst) {
  __shared__ _Float16 tile[32][33];
  const int b = blockIdx.z;
  const int h0 = blockIdx.x * 32, c0 = blockIdx.y * 32;
  const float* s = src + (size_t)b * CC * HWD;
  _Float16* d = dst + (size_t)b * HWD * CC;
#pragma unroll
  for (int i = threadIdx.y; i < 32; i += 8)
    tile[i][threadIdx.x] = (_Float16)s[(size_t)(c0 + i) * HWD + h0 + threadIdx.x];
  __syncthreads();
#pragma unroll
  for (int i = threadIdx.y; i < 32; i += 8)
    d[(size_t)(h0 + i) * CC + c0 + threadIdx.x] = tile[threadIdx.x][i];
}

// ---------------- GEMM1: xT[b][o][c] = sum_h Wc[o][h] * n1[b][c][h] ----------------
// A = Wc rows (per-wave unique, kept in regs from global); B panel (shared by all
// 8 waves) staged in LDS via async copies, double-buffered. Per k-step: burst-load
// all 8 B-fragments from LDS, then run the 8-WMMA chain uninterrupted.
__global__ __launch_bounds__(256) void gemm1_xT(const _Float16* __restrict__ wch,
                                                const _Float16* __restrict__ n1h,
                                                _Float16* __restrict__ xT) {
  __shared__ _Float16 bpan[2][128 * LPITCH];   // 2 x 10KB
  const int tid  = threadIdx.x;
  const int lane = tid & 31;
  const int wv   = tid >> 5;                     // 0..7
  const int b    = blockIdx.z;
  const int ob   = blockIdx.x * 128 + wv * 16;   // output rows (o)
  const int cb   = blockIdx.y * 128;             // output cols (c)

  const _Float16* n1b = n1h + (size_t)b * CC * HWD + (size_t)cb * HWD;
  _Float16* xTb = xT + (size_t)b * HWD * CC;

  stage_panel128(n1b, HWD, bpan[0], tid);

  v8f acc[8] = {};
  int pb = 0;
  for (int kk = 0; kk < HWD; kk += 32, pb ^= 1) {
    wait_async0();
    __syncthreads();                             // panel pb ready; buffer pb^1 free
    if (kk + 32 < HWD)
      stage_panel128(n1b + kk + 32, HWD, bpan[pb ^ 1], tid);
    __builtin_prefetch(wch + (size_t)ob * HWD + kk + 64, 0, 1);
    v16h a = load_frag_a(wch + (size_t)ob * HWD + kk, HWD, lane);
    v16h bf[8];
#pragma unroll
    for (int t = 0; t < 8; ++t)
      bf[t] = lds_frag_b(bpan[pb] + t * 16 * LPITCH, lane);
#pragma unroll
    for (int t = 0; t < 8; ++t)
      acc[t] = __builtin_amdgcn_wmma_f32_16x16x32_f16(false, a, false, bf[t],
                                                      (short)0, acc[t], false, false);
  }

  const int sel = lane >> 4, col = lane & 15;
#pragma unroll
  for (int t = 0; t < 8; ++t)
#pragma unroll
    for (int v = 0; v < 8; ++v)
      xTb[(size_t)(ob + v + 8 * sel) * CC + cb + t * 16 + col] = (_Float16)acc[t][v];
}

// ---------------- GEMM2 + softmax: e[o][h] = sum_c xT[o][c]*n2T[h][c];
// a = softmax_h(e); write aT[b][h][o] (f16) so GEMM3 B-frags are contiguous.
// The A tile (shared by all 4 waves) is staged in LDS via async copies. ----------------
__global__ __launch_bounds__(128) void attn_softmax(const _Float16* __restrict__ xT,
                                                    const _Float16* __restrict__ n2T,
                                                    _Float16* __restrict__ aT) {
  __shared__ _Float16 apan[2][16 * LPITCH];      // 2 x 1.25KB
  const int tid  = threadIdx.x;
  const int lane = tid & 31;
  const int wv   = tid >> 5;                     // 0..3, each wave owns 256 h-columns
  const int b    = blockIdx.y;
  const int ob   = blockIdx.x * 16;              // 16 o-rows per workgroup

  const _Float16* xA  = xT  + (size_t)b * HWD * CC + (size_t)ob * CC;
  const _Float16* nB  = n2T + (size_t)b * HWD * CC + (size_t)(wv * 256) * CC;
  _Float16* aTb = aT + (size_t)b * HWD * HWD;

  stage_tile16(xA, CC, apan[0], tid);

  v8f acc[16] = {};
  int pb = 0;
  for (int kc = 0; kc < CC; kc += 32, pb ^= 1) {
    wait_async0();
    __syncthreads();
    if (kc + 32 < CC)
      stage_tile16(xA + kc + 32, CC, apan[pb ^ 1], tid);
    v16h a = lds_frag_a(apan[pb], lane);
    // 4 groups of 4: burst the global B-frag loads, then the WMMA chain.
#pragma unroll
    for (int g = 0; g < 4; ++g) {
      v16h bf[4];
#pragma unroll
      for (int u = 0; u < 4; ++u)
        bf[u] = load_frag_b(nB + (size_t)((g * 4 + u) * 16) * CC + kc, CC, lane);
#pragma unroll
      for (int u = 0; u < 4; ++u)
        acc[g * 4 + u] = __builtin_amdgcn_wmma_f32_16x16x32_f16(
            false, a, false, bf[u], (short)0, acc[g * 4 + u], false, false);
    }
  }

  // ---- row softmax over h (N dim spread over 4 waves x 16 tiles x 16 lanes) ----
  __shared__ float red[4][32][8];
  __shared__ float rowmax[16];
  __shared__ float rowsum[16];
  const int sel = lane >> 4;

  float pm[8];
#pragma unroll
  for (int v = 0; v < 8; ++v) pm[v] = -3.402823466e38f;
#pragma unroll
  for (int t = 0; t < 16; ++t)
#pragma unroll
    for (int v = 0; v < 8; ++v) pm[v] = fmaxf(pm[v], acc[t][v]);
#pragma unroll
  for (int v = 0; v < 8; ++v) red[wv][lane][v] = pm[v];
  __syncthreads();

  if (tid < 16) {
    const int M = tid, g = (M >> 3) * 16, v = M & 7;
    float m = -3.402823466e38f;
    for (int w = 0; w < 4; ++w)
      for (int l = g; l < g + 16; ++l) m = fmaxf(m, red[w][l][v]);
    rowmax[M] = m;
  }
  __syncthreads();

  float rmax[8], ps[8];
#pragma unroll
  for (int v = 0; v < 8; ++v) { rmax[v] = rowmax[v + 8 * sel]; ps[v] = 0.f; }
#pragma unroll
  for (int t = 0; t < 16; ++t)
#pragma unroll
    for (int v = 0; v < 8; ++v) {
      float e = __expf(acc[t][v] - rmax[v]);
      acc[t][v] = e;
      ps[v] += e;
    }
#pragma unroll
  for (int v = 0; v < 8; ++v) red[wv][lane][v] = ps[v];
  __syncthreads();

  if (tid < 16) {
    const int M = tid, g = (M >> 3) * 16, v = M & 7;
    float s = 0.f;
    for (int w = 0; w < 4; ++w)
      for (int l = g; l < g + 16; ++l) s += red[w][l][v];
    rowsum[M] = s;
  }
  __syncthreads();

  float rinv[8];
#pragma unroll
  for (int v = 0; v < 8; ++v) rinv[v] = 1.0f / rowsum[v + 8 * sel];

  // aT[h][o]: tile t column h = wv*256 + t*16 + (lane&15); 8 consecutive o per lane -> b128 stores
#pragma unroll
  for (int t = 0; t < 16; ++t) {
    v8h o8;
#pragma unroll
    for (int v = 0; v < 8; ++v) o8[v] = (_Float16)(acc[t][v] * rinv[v]);
    const int N = wv * 256 + t * 16 + (lane & 15);
    *(v8h*)(aTb + (size_t)N * HWD + ob + 8 * sel) = o8;
  }
}

// ---------------- GEMM3: m[b][c][j] = sum_k n2[b][c][k] * aT[b][j][k] ----------------
__global__ __launch_bounds__(256) void gemm3_out(const _Float16* __restrict__ n2h,
                                                 const _Float16* __restrict__ aT,
                                                 float* __restrict__ out) {
  __shared__ _Float16 bpan[2][128 * LPITCH];     // 2 x 10KB
  const int tid  = threadIdx.x;
  const int lane = tid & 31;
  const int wv   = tid >> 5;                     // 0..7
  const int b    = blockIdx.z;
  const int cbse = blockIdx.x * 128 + wv * 16;   // output rows (c)
  const int jbse = blockIdx.y * 128;             // output cols (j)

  const _Float16* A  = n2h + (size_t)b * CC * HWD + (size_t)cbse * HWD;
  const _Float16* Bp = aT  + (size_t)b * HWD * HWD + (size_t)jbse * HWD;
  float* ob = out + (size_t)b * CC * HWD;

  stage_panel128(Bp, HWD, bpan[0], tid);

  v8f acc[8] = {};
  int pb = 0;
  for (int kk = 0; kk < HWD; kk += 32, pb ^= 1) {
    wait_async0();
    __syncthreads();
    if (kk + 32 < HWD)
      stage_panel128(Bp + kk + 32, HWD, bpan[pb ^ 1], tid);
    __builtin_prefetch(A + kk + 64, 0, 1);
    v16h a = load_frag_a(A + kk, HWD, lane);
    v16h bf[8];
#pragma unroll
    for (int t = 0; t < 8; ++t)
      bf[t] = lds_frag_b(bpan[pb] + t * 16 * LPITCH, lane);
#pragma unroll
    for (int t = 0; t < 8; ++t)
      acc[t] = __builtin_amdgcn_wmma_f32_16x16x32_f16(false, a, false, bf[t],
                                                      (short)0, acc[t], false, false);
  }

  const int sel = lane >> 4, col = lane & 15;
#pragma unroll
  for (int t = 0; t < 8; ++t)
#pragma unroll
    for (int v = 0; v < 8; ++v)
      ob[(size_t)(cbse + v + 8 * sel) * HWD + jbse + t * 16 + col] = acc[t][v];
}

extern "C" void kernel_launch(void* const* d_in, const int* in_sizes, int n_in,
                              void* d_out, int out_size, void* d_ws, size_t ws_size,
                              hipStream_t stream) {
  (void)in_sizes; (void)n_in; (void)out_size; (void)ws_size;
  const float* n1 = (const float*)d_in[0];   // [B][C][HW]
  const float* n2 = (const float*)d_in[1];   // [B][C][HW]
  const float* Wc = (const float*)d_in[2];   // [HW][HW]
  float* out = (float*)d_out;                // [B][C][HW]

  const size_t nBCHW = (size_t)BB * CC * HWD;          // 8,388,608
  const size_t nWC   = (size_t)HWD * HWD;              // 1,048,576
  const size_t nA    = (size_t)BB * HWD * HWD;         // 16,777,216

  char* ws = (char*)d_ws;
  size_t off = 0;
  auto alloc = [&](size_t bytes) -> void* {
    void* p = ws + off;
    off += (bytes + 255) & ~(size_t)255;
    return p;
  };
  _Float16* n1h  = (_Float16*)alloc(nBCHW * 2);  // n1 in f16
  _Float16* n2h  = (_Float16*)alloc(nBCHW * 2);  // n2 in f16
  _Float16* n2T  = (_Float16*)alloc(nBCHW * 2);  // n2 transposed [B][HW][C] f16
  _Float16* wch  = (_Float16*)alloc(nWC * 2);    // W_c in f16
  _Float16* xT   = (_Float16*)alloc(nBCHW * 2);  // x transposed [B][HW][C] f16
  _Float16* aTm  = (_Float16*)alloc(nA * 2);     // softmax(e) transposed [B][HW_j][HW_k] f16

  // Stage 0: precision conversion + transpose (bandwidth-bound, trivially fast)
  convert_f32_f16<<<2048, 256, 0, stream>>>(n1, n1h, nBCHW / 4);
  convert_f32_f16<<<2048, 256, 0, stream>>>(n2, n2h, nBCHW / 4);
  convert_f32_f16<<<512,  256, 0, stream>>>(Wc, wch, nWC / 4);
  transpose_convert<<<dim3(HWD / 32, CC / 32, BB), dim3(32, 8), 0, stream>>>(n2, n2T);

  // Stage 1: xT = (n1 @ Wc^T)^T
  gemm1_xT<<<dim3(HWD / 128, CC / 128, BB), 256, 0, stream>>>(wch, n1h, xT);

  // Stage 2: e = xT @ n2T^T ; aT = softmax_rows(e)^T  (fused)
  attn_softmax<<<dim3(HWD / 16, BB), 128, 0, stream>>>(xT, n2T, aTm);

  // Stage 3: m = n2 @ a
  gemm3_out<<<dim3(CC / 128, HWD / 128, BB), 256, 0, stream>>>(n2h, aTm, out);
}
